// TrajectoryAutoencoder_9216999817930
// MI455X (gfx1250) — matrix operand order repaired and checked by reading, not compiled, not run on hardware
//
#include <hip/hip_runtime.h>

// MI455X / gfx1250, wave32. WMMA f16 (16x16x32, f32 accum) for all matmuls in
// the recurrence; f32 everywhere state/precision matters. Tensor Data Mover
// (async tensor_load_to_lds) streams per-step weights into LDS, overlapped
// with the WMMA matvec.
//
// Workspace layout (requires ws_size >= ~10 MB):
//   [0      , 4 MiB)   encPre  f32 [512][2048]  = traj@enc_Wi + enc_bh
//   [4 MiB  , 6 MiB)   encFrag f16 [16][128][32][16]  (Wh in WMMA-B layout)
//   [6 MiB  , 8.5 MiB) decFrag f16 [20][128][32][16]  ([Wi;Wh] concat, B layout)
//   [9 MiB  , +1 KiB)  embedding f32 [256]

typedef __attribute__((ext_vector_type(16))) _Float16 v16h;
typedef __attribute__((ext_vector_type(8)))  float    v8f;
typedef __attribute__((ext_vector_type(4)))  unsigned v4u;
typedef __attribute__((ext_vector_type(8)))  int      v8i;
typedef __attribute__((ext_vector_type(4)))  int      v4i;

namespace {
constexpr int kT = 512, kD = 128, kH = 512, kG = 2048, kEMB = 256;
constexpr size_t ENC_FRAG_OFF = size_t(4) << 20;
constexpr size_t DEC_FRAG_OFF = size_t(6) << 20;
constexpr size_t EMB_OFF      = size_t(9) << 20;

__device__ __forceinline__ float sigmoidf(float x) {
  return 1.0f / (1.0f + __expf(-x));
}
__device__ __forceinline__ v8f wmma_f16(v16h a, v16h b, v8f c) {
  // emits v_wmma_f32_16x16x32_f16
  return __builtin_amdgcn_wmma_f32_16x16x32_f16(false, a, false, b,
                                                (short)0, c, false, false);
}

// TDM: async DMA of `nelem` contiguous f32 (nelem <= 65535) from global to
// LDS. D# fields per CDNA5 ISA ch.8: group0 {count=1, lds_addr, global_addr,
// type=2}; group1 {data_size=4B, tensor_dim0=nelem, tile_dim0=nelem,
// tensor_dim0_stride=nelem}; groups 2/3 zero (<=2D tensor). Tracked by
// TENSORcnt; pair with s_wait_tensorcnt.
__device__ __forceinline__ void tdm_load_f32(unsigned ldsAddr, const void* gptr,
                                             unsigned nelem) {
  const unsigned long long ga = (unsigned long long)(uintptr_t)gptr;
  v4u g0;
  g0[0] = 1u;                                            // count=1 (valid)
  g0[1] = ldsAddr;                                       // lds_addr
  g0[2] = (unsigned)ga;                                  // global_addr[31:0]
  g0[3] = (unsigned)((ga >> 32) & 0x01FFFFFFull) | (2u << 30);  // [56:32]|type=2
  v8i g1;
  g1[0] = (int)(2u << 16);                    // data_size = 4 bytes
  g1[1] = (int)((nelem & 0xffffu) << 16);     // tensor_dim0[15:0]
  g1[2] = (int)((nelem >> 16) | (1u << 16));  // tensor_dim0[31:16], dim1=1
  g1[3] = (int)((nelem & 0xffffu) << 16);     // tile_dim0 = nelem
  g1[4] = 0;                                  // tile_dim1/2 unused (1-D tile)
  g1[5] = (int)nelem;                         // tensor_dim0_stride lo32
  g1[6] = 0;
  g1[7] = 0;
  v4i gz = {};
#if defined(__clang_major__) && __clang_major__ >= 23
  v8i gz8 = {};
  __builtin_amdgcn_tensor_load_to_lds(g0, g1, gz, gz, gz8, 0);
#else
  __builtin_amdgcn_tensor_load_to_lds(g0, g1, gz, gz, 0);
#endif
}
}  // namespace

// ---------------------------------------------------------------------------
// Kernel 1: encPre[t][n] = sum_d traj[t][d]*enc_Wi[d][n] + enc_bh[n]
// 512x2048x128 GEMM; one 16x16 output tile per wave, K in 4 chunks of 32.
// A layout (16-bit 16x32): lane(m=L&15,hi=L>>4), elem e -> k = e+hi*16
// B layout (16-bit 32x16): lane(n=L&15,hi=L>>4), elem e -> k = e+hi*16
// D layout (f32 16x16):    lane(n=L&15,hi),     vgpr r -> m = r+hi*8
// ---------------------------------------------------------------------------
__global__ void __launch_bounds__(256)
enc_pre_gemm(const float* __restrict__ traj, const float* __restrict__ Wi,
             const float* __restrict__ bh, float* __restrict__ encPre) {
  const int lane = threadIdx.x & 31;
  const int wave = threadIdx.x >> 5;
  const int w = blockIdx.x * 8 + wave;  // 0..4095 tiles
  const int ttile = w >> 7, ntile = w & 127;
  const int low = lane & 15, hi = lane >> 4;

  v8f acc = {};
#pragma unroll
  for (int kc = 0; kc < 4; ++kc) {
    v16h a, b;
#pragma unroll
    for (int e = 0; e < 16; ++e) {
      const int ka = kc * 32 + e + hi * 16;          // < 128
      a[e] = (_Float16)traj[(ttile * 16 + low) * kD + ka];
      const int kb = kc * 32 + e + hi * 16;
      b[e] = (_Float16)Wi[kb * kG + ntile * 16 + low];
    }
    acc = wmma_f16(a, b, acc);
  }
#pragma unroll
  for (int r = 0; r < 8; ++r) {
    const int t = ttile * 16 + r + hi * 8;
    const int n = ntile * 16 + low;
    encPre[(size_t)t * kG + n] = acc[r] + bh[n];
  }
}

// ---------------------------------------------------------------------------
// Kernel 2: pack recurrent weights (f32 -> f16) into WMMA B-fragment layout.
// frag linear index = ((kc*128 + nt)*32 + lane)*16 + e
// encFrag: K=512 from enc_Wh.   decFrag: K=640 = [dec_Wi(128) ; dec_Wh(512)].
// ---------------------------------------------------------------------------
__global__ void __launch_bounds__(256)
build_frags(const float* __restrict__ encWh, const float* __restrict__ decWi,
            const float* __restrict__ decWh, _Float16* __restrict__ encFrag,
            _Float16* __restrict__ decFrag) {
  const unsigned idx = blockIdx.x * 256u + threadIdx.x;
  const unsigned encCount = 16u * 128u * 32u * 16u;  // 1,048,576
  if (idx < encCount) {
    const unsigned e = idx & 15, lane = (idx >> 4) & 31, nt = (idx >> 9) & 127,
                   kc = idx >> 16;
    const unsigned n = nt * 16 + (lane & 15);
    const unsigned k = kc * 32 + e + (lane >> 4) * 16;
    encFrag[idx] = (_Float16)encWh[(size_t)k * kG + n];
  } else {
    const unsigned j = idx - encCount;
    if (j >= 20u * 128u * 32u * 16u) return;
    const unsigned e = j & 15, lane = (j >> 4) & 31, nt = (j >> 9) & 127,
                   kc = j >> 16;
    const unsigned n = nt * 16 + (lane & 15);
    const unsigned k = kc * 32 + e + (lane >> 4) * 16;
    const float v = (k < 128) ? decWi[(size_t)k * kG + n]
                              : decWh[(size_t)(k - 128) * kG + n];
    decFrag[j] = (_Float16)v;
  }
}

// ---------------------------------------------------------------------------
// Kernel 3: encoder recurrence (single WG, 32 waves). Each step:
//   - each wave TDM-loads its 64-float slice of encPre[t] into LDS (async),
//   - z[2048] = h @ Wh via WMMA (M=1 row in A) overlapping the DMA,
//   - gates in f32, h/c update. Each wave owns 4 N-tiles; 16 K-chunks of 32.
// ---------------------------------------------------------------------------
__global__ void __launch_bounds__(1024)
encoder_rnn(const float* __restrict__ encPre,
            const _Float16* __restrict__ encFrag,
            const float* __restrict__ embW, const float* __restrict__ embB,
            float* __restrict__ embOut, float* __restrict__ out) {
  __shared__ float h32[kH];
  __shared__ _Float16 h16[kH];
  __shared__ float zsh[kG];
  __shared__ float preLds[kG];

  const int tid = threadIdx.x, lane = tid & 31, wave = tid >> 5;
  const int low = lane & 15, hi = lane >> 4;
  const unsigned waveU = (unsigned)__builtin_amdgcn_readfirstlane(wave);
  const unsigned preLdsBase = (unsigned)(uintptr_t)(&preLds[0]);

  float creg = 0.0f;
  if (tid < kH) { h32[tid] = 0.0f; h16[tid] = (_Float16)0.0f; }
  __syncthreads();

  for (int t = 0; t < kT; ++t) {
    // Async: stage this step's precomputed gate row (8 KB) into LDS.
    tdm_load_f32(preLdsBase + waveU * 256u,
                 encPre + (size_t)t * kG + waveU * 64u, 64u);

    v8f a0 = {}, a1 = {}, a2 = {}, a3 = {};
    const int ntbase = wave * 4;
    for (int kc = 0; kc < 16; ++kc) {
      v16h a;
#pragma unroll
      for (int e = 0; e < 16; ++e) {
        const int ka = kc * 32 + e + hi * 16;  // < 512
        a[e] = (low == 0) ? h16[ka] : (_Float16)0.0f;   // only row m=0 live
      }
      const v16h* bp = reinterpret_cast<const v16h*>(encFrag) +
                       (size_t)(kc * 128 + ntbase) * 32 + lane;
      a0 = wmma_f16(a, bp[0], a0);
      a1 = wmma_f16(a, bp[32], a1);
      a2 = wmma_f16(a, bp[64], a2);
      a3 = wmma_f16(a, bp[96], a3);
    }
    if (lane < 16) {  // row m=0 lives in vgpr0 of lanes 0..15
      zsh[(ntbase + 0) * 16 + lane] = a0[0];
      zsh[(ntbase + 1) * 16 + lane] = a1[0];
      zsh[(ntbase + 2) * 16 + lane] = a2[0];
      zsh[(ntbase + 3) * 16 + lane] = a3[0];
    }
    __builtin_amdgcn_s_wait_tensorcnt(0);  // own slice landed
    __syncthreads();                       // all slices + z visible
    if (tid < kH) {
      const float zi = zsh[tid] + preLds[tid];
      const float zf = zsh[kH + tid] + preLds[kH + tid];
      const float zg = zsh[2 * kH + tid] + preLds[2 * kH + tid];
      const float zo = zsh[3 * kH + tid] + preLds[3 * kH + tid];
      creg = sigmoidf(zf) * creg + sigmoidf(zi) * tanhf(zg);
      const float hn = sigmoidf(zo) * tanhf(creg);
      h32[tid] = hn;
      h16[tid] = (_Float16)hn;
    }
    __syncthreads();
  }

  if (tid < kEMB) {  // embedding = h @ enc_emb_W + b (exact f32, one-shot)
    float acc = embB[tid];
    for (int k = 0; k < kH; ++k) acc += h32[k] * embW[k * kEMB + tid];
    embOut[tid] = acc;
    out[kT * kD + tid] = acc;  // second tuple output
  }
}

// ---------------------------------------------------------------------------
// Kernel 4: decoder recurrence. z = [x;h] @ [Wi;Wh] (20 K-chunks) via WMMA.
// Per-step output dense (512x128, 256 KB): each wave TDM-loads a 2048-float
// slice into dynamic LDS at step start; the DMA overlaps the WMMA matvec and
// the gate phase; projection then runs entirely out of LDS in exact f32.
// ---------------------------------------------------------------------------
__global__ void __launch_bounds__(1024)
decoder_rnn(const float* __restrict__ emb, const float* __restrict__ initW,
            const float* __restrict__ initB,
            const _Float16* __restrict__ decFrag,
            const float* __restrict__ decBh, const float* __restrict__ outW,
            const float* __restrict__ outB, float* __restrict__ out) {
  __shared__ float h32[kH];
  __shared__ _Float16 xh16[kD + kH];  // [0:128)=x, [128:640)=h
  __shared__ float zsh[kG];
  __shared__ float red[1024];
  extern __shared__ float wtLds[];    // 65536 f32 = 256 KB (dynamic)

  const int tid = threadIdx.x, lane = tid & 31, wave = tid >> 5;
  const int low = lane & 15, hi = lane >> 4;
  const unsigned waveU = (unsigned)__builtin_amdgcn_readfirstlane(wave);
  const unsigned wtLdsBase = (unsigned)(uintptr_t)(&wtLds[0]);

  float creg = 0.0f;
  if (tid < kH) {  // hd0 = embedding @ dec_init_W + b
    float acc = initB[tid];
    for (int e = 0; e < kEMB; ++e) acc += emb[e] * initW[e * kH + tid];
    h32[tid] = acc;
    xh16[kD + tid] = (_Float16)acc;
  }
  if (tid < kD) xh16[tid] = (_Float16)0.0f;  // x0 = 0
  __syncthreads();

  for (int t = 0; t < kT; ++t) {
    // Async: each wave DMAs its 8 KB slice of outW[t] (rows 16w..16w+15).
    tdm_load_f32(wtLdsBase + waveU * 8192u,
                 outW + (size_t)t * kH * kD + waveU * 2048u, 2048u);

    v8f a0 = {}, a1 = {}, a2 = {}, a3 = {};
    const int ntbase = wave * 4;
    for (int kc = 0; kc < 20; ++kc) {
      v16h a;
#pragma unroll
      for (int e = 0; e < 16; ++e) {
        const int ka = kc * 32 + e + hi * 16;  // < 640
        a[e] = (low == 0) ? xh16[ka] : (_Float16)0.0f;
      }
      const v16h* bp = reinterpret_cast<const v16h*>(decFrag) +
                       (size_t)(kc * 128 + ntbase) * 32 + lane;
      a0 = wmma_f16(a, bp[0], a0);
      a1 = wmma_f16(a, bp[32], a1);
      a2 = wmma_f16(a, bp[64], a2);
      a3 = wmma_f16(a, bp[96], a3);
    }
    if (lane < 16) {
      zsh[(ntbase + 0) * 16 + lane] = a0[0];
      zsh[(ntbase + 1) * 16 + lane] = a1[0];
      zsh[(ntbase + 2) * 16 + lane] = a2[0];
      zsh[(ntbase + 3) * 16 + lane] = a3[0];
    }
    __syncthreads();
    if (tid < kH) {
      const float zi = zsh[tid] + decBh[tid];
      const float zf = zsh[kH + tid] + decBh[kH + tid];
      const float zg = zsh[2 * kH + tid] + decBh[2 * kH + tid];
      const float zo = zsh[3 * kH + tid] + decBh[3 * kH + tid];
      creg = sigmoidf(zf) * creg + sigmoidf(zi) * tanhf(zg);
      const float hn = sigmoidf(zo) * tanhf(creg);
      h32[tid] = hn;
      xh16[kD + tid] = (_Float16)hn;
    }
    __builtin_amdgcn_s_wait_tensorcnt(0);  // own DMA slice landed
    __syncthreads();                       // all slices + h visible
    {  // x = h @ outW[t] + outB[t]  (8 threads per output column, LDS only)
      const int dcol = tid & 127, seg = tid >> 7, kb = seg * 64;
      float p = 0.0f;
      for (int kk = 0; kk < 64; ++kk)
        p += h32[kb + kk] * wtLds[(kb + kk) * kD + dcol];
      red[tid] = p;  // tid == seg*128 + dcol
    }
    __syncthreads();
    if (tid < kD) {
      float s = outB[t * kD + tid];
#pragma unroll
      for (int seg = 0; seg < 8; ++seg) s += red[seg * kD + tid];
      out[t * kD + tid] = s;          // reconstruction[t]
      xh16[tid] = (_Float16)s;        // feeds next step
    }
    __syncthreads();
  }
}

extern "C" void kernel_launch(void* const* d_in, const int* in_sizes, int n_in,
                              void* d_out, int out_size, void* d_ws,
                              size_t ws_size, hipStream_t stream) {
  const float* traj  = (const float*)d_in[0];
  const float* encWi = (const float*)d_in[1];
  const float* encWh = (const float*)d_in[2];
  const float* encBh = (const float*)d_in[3];
  const float* embW  = (const float*)d_in[4];
  const float* embB  = (const float*)d_in[5];
  const float* initW = (const float*)d_in[6];
  const float* initB = (const float*)d_in[7];
  const float* decWi = (const float*)d_in[8];
  const float* decWh = (const float*)d_in[9];
  const float* decBh = (const float*)d_in[10];
  const float* outW  = (const float*)d_in[11];
  const float* outB  = (const float*)d_in[12];
  float* out = (float*)d_out;

  char* ws = (char*)d_ws;
  float* encPre = (float*)ws;
  _Float16* encFrag = (_Float16*)(ws + ENC_FRAG_OFF);
  _Float16* decFrag = (_Float16*)(ws + DEC_FRAG_OFF);
  float* embBuf = (float*)(ws + EMB_OFF);

  // Opt in to 256 KB dynamic LDS for the decoder (ISA: up to 320 KB per WG).
  (void)hipFuncSetAttribute(reinterpret_cast<const void*>(decoder_rnn),
                            hipFuncAttributeMaxDynamicSharedMemorySize,
                            kH * kD * (int)sizeof(float));

  enc_pre_gemm<<<dim3(512), dim3(256), 0, stream>>>(traj, encWi, encBh, encPre);
  build_frags<<<dim3(9216), dim3(256), 0, stream>>>(encWh, decWi, decWh,
                                                    encFrag, decFrag);
  encoder_rnn<<<dim3(1), dim3(1024), 0, stream>>>(encPre, encFrag, embW, embB,
                                                  embBuf, out);
  decoder_rnn<<<dim3(1), dim3(1024), kH * kD * (int)sizeof(float), stream>>>(
      embBuf, initW, initB, decFrag, decBh, outW, outB, out);
}
